// GraphAttentionLayer_76991583748612
// MI455X (gfx1250) — matrix operand order repaired and checked by reading, not compiled, not run on hardware
//
#include <hip/hip_runtime.h>
#include <hip/hip_bf16.h>
#include <math.h>

// GAT layer: B=8, N=1024, F=256, H=4, U=256
#define B_  8
#define N_  1024
#define F_  256
#define H_  4
#define U_  256

typedef __bf16 bf16_t;
typedef bf16_t bf16x16 __attribute__((ext_vector_type(16)));
typedef bf16_t bf16x8  __attribute__((ext_vector_type(8)));
typedef float  f32x8   __attribute__((ext_vector_type(8)));

static_assert(sizeof(bf16x16) == 32, "bf16x16 must be 32 bytes");
static_assert(sizeof(bf16x8) == 16, "bf16x8 must be 16 bytes");

#if defined(__has_builtin)
#  if __has_builtin(__builtin_amdgcn_global_load_async_to_lds_b32)
#    define GAT_ASYNC_LDS 1
#  endif
#endif

// ---------------------------------------------------------------------------
// Kernel 1: h = x @ W per head (bf16 WMMA, f32 accum).
//   - stores h as bf16, layout [bh][u][n]  (u-major, so kernel2 B-tiles are
//     contiguous 32B per lane)
//   - computes src/dst logits from the *f32* accumulators (accuracy for exp)
// Block: 256 thr (8 waves) computes 64 rows x 256 cols of h for one (b,head).
// Wave w: row-tile rt=w&3 (16 rows), col-group cg=w>>2 (128 cols = 8 u-tiles).
// W^T slice is double-buffered in LDS: one barrier per K-step.
// ---------------------------------------------------------------------------
__global__ __launch_bounds__(256) void gat_proj(
    const float* __restrict__ x, const float* __restrict__ W,
    const float* __restrict__ a_src, const float* __restrict__ a_dst,
    bf16_t* __restrict__ hbf, float* __restrict__ srcv, float* __restrict__ dstv)
{
  __shared__ bf16_t WT[2 * U_ * 32];    // double-buffered W^T slice [u][k], 2x16KB
  __shared__ float  ldsS[64], ldsD[64]; // src/dst row accumulators

  const int tid  = threadIdx.x;
  const int lane = tid & 31;
  const int w    = tid >> 5;
  const int rt   = w & 3;
  const int cg   = w >> 2;

  const int blk  = blockIdx.x;          // B*H*(N/64) = 512
  const int nt   = blk & 15;
  const int bh   = blk >> 4;            // b*H + head
  const int head = bh & (H_ - 1);
  const int b    = bh >> 2;
  const int n0   = nt * 64;

  if (tid < 64) { ldsS[tid] = 0.f; ldsD[tid] = 0.f; }

  const int halfK = (lane >> 4) * 8;    // unused here; A tile also uses kl
  const int kl    = (lane >> 4) * 16;   // B-layout K base per lane half
  const int row   = n0 + rt * 16 + (lane & 15);
  const float* xrow = x + ((size_t)b * N_ + row) * F_;
  const float* Wh   = W + (size_t)head * F_ * U_;

  f32x8 c[8] = {};                      // 8 u-tiles of 16x16 f32 accum

  for (int kk = 0; kk < F_ / 32; ++kk) {
    const int k0 = kk * 32;
    bf16_t* WTb = WT + (kk & 1) * (U_ * 32);

    // stage W^T (bf16) into LDS.  Thread tid owns column u=tid: 32 global
    // loads off one base pointer with immediate offsets (k*1024B), converted
    // with v_cvt_pk_bf16_f32, then one contiguous 64B LDS row (2x 32B store).
    // For fixed k, consecutive tid -> consecutive u: fully coalesced.
    {
      const float* wp = Wh + (size_t)k0 * U_ + tid;
      bf16x16 t0, t1;
#pragma unroll
      for (int k2 = 0; k2 < 16; ++k2) t0[k2] = (bf16_t)wp[k2 * U_];
#pragma unroll
      for (int k2 = 0; k2 < 16; ++k2) t1[k2] = (bf16_t)wp[(16 + k2) * U_];
      *(bf16x16*)&WTb[tid * 32]      = t0;
      *(bf16x16*)&WTb[tid * 32 + 16] = t1;
    }
    __syncthreads();

    // A tile (16x32 bf16): lane holds row M=lane&15; element e -> K =
    // kl + e   (contiguous, matches B)
    bf16x16 a;
    {
      const float4* p0 = (const float4*)(xrow + k0 + kl);
      const float4* p1 = (const float4*)(xrow + k0 + kl + 8);
      float4 r0 = p0[0], r1 = p0[1], r2 = p1[0], r3 = p1[1];
      a[0]=(bf16_t)r0.x;  a[1]=(bf16_t)r0.y;  a[2]=(bf16_t)r0.z;  a[3]=(bf16_t)r0.w;
      a[4]=(bf16_t)r1.x;  a[5]=(bf16_t)r1.y;  a[6]=(bf16_t)r1.z;  a[7]=(bf16_t)r1.w;
      a[8]=(bf16_t)r2.x;  a[9]=(bf16_t)r2.y;  a[10]=(bf16_t)r2.z; a[11]=(bf16_t)r2.w;
      a[12]=(bf16_t)r3.x; a[13]=(bf16_t)r3.y; a[14]=(bf16_t)r3.z; a[15]=(bf16_t)r3.w;
    }

    // preload all 8 B tiles into distinct registers so the ds_loads pipeline
    // and the 8 WMMAs can issue back-to-back
    bf16x16 bm[8];
#pragma unroll
    for (int ut = 0; ut < 8; ++ut) {
      const int u = cg * 128 + ut * 16 + (lane & 15);
      bm[ut] = *(const bf16x16*)&WTb[u * 32 + kl];
    }
#pragma unroll
    for (int ut = 0; ut < 8; ++ut)
      c[ut] = __builtin_amdgcn_wmma_f32_16x16x32_bf16(
          false, a, false, bm[ut], (short)0, c[ut], false, false);
  }

  // src/dst logits from f32 accumulators (C layout: elem v -> M=(lane>>4)*8+v,
  // N-col = lane&15); reduce over u via LDS float atomics.
  float av[8], dv[8];
#pragma unroll
  for (int ut = 0; ut < 8; ++ut) {
    const int u = cg * 128 + ut * 16 + (lane & 15);
    av[ut] = a_src[head * U_ + u];
    dv[ut] = a_dst[head * U_ + u];
  }
#pragma unroll
  for (int v = 0; v < 8; ++v) {
    float ps = 0.f, pd = 0.f;
#pragma unroll
    for (int ut = 0; ut < 8; ++ut) { float cv = c[ut][v]; ps += cv * av[ut]; pd += cv * dv[ut]; }
    const int rl = rt * 16 + (lane >> 4) * 8 + v;
    atomicAdd(&ldsS[rl], ps);
    atomicAdd(&ldsD[rl], pd);
  }

  // store h as bf16 [bh][u][n]; the 8 accum elems per u-tile are contiguous
  // in n -> one 16B global_store_b128 each
  const int nbase = n0 + rt * 16 + (lane >> 4) * 8;
#pragma unroll
  for (int ut = 0; ut < 8; ++ut) {
    const int u = cg * 128 + ut * 16 + (lane & 15);
    bf16x8 hv;
#pragma unroll
    for (int v = 0; v < 8; ++v) hv[v] = (bf16_t)c[ut][v];
    *(bf16x8*)(hbf + ((size_t)bh * U_ + u) * N_ + nbase) = hv;
  }

  __syncthreads();
  if (tid < 64)       srcv[bh * N_ + n0 + tid]        = ldsS[tid];
  else if (tid < 128) dstv[bh * N_ + n0 + (tid - 64)] = ldsD[tid - 64];
}

// ---------------------------------------------------------------------------
// Kernel 2: flash-style attention.  scores=lrelu(src_i+dst_j); since lrelu is
// monotonic, exact row max m_i = lrelu(src_i + max_j dst_j) -> no rescaling.
// O_i = (1/Z_i) * sum_j exp(lrelu(src_i+dst_j)-m_i) * h_j   via bf16 WMMA.
// Block: 256 thr, one (b,head, 32-row i-tile). Wave: isub=w&1 (16 rows),
// ug=w>>1 (64 u cols = 4 tiles).
// ---------------------------------------------------------------------------
__global__ __launch_bounds__(256) void gat_attn(
    const bf16_t* __restrict__ hbf, const float* __restrict__ srcv,
    const float* __restrict__ dstv, float* __restrict__ out)
{
  __shared__ float ldsDst[N_];   // 4 KB: dst logits for this (b,head)
  __shared__ float red[256];
  __shared__ float ldsZ[32];

  const int tid  = threadIdx.x;
  const int lane = tid & 31;
  const int w    = tid >> 5;
  const int isub = w & 1;
  const int ug   = w >> 1;

  const int blk  = blockIdx.x;          // B*H*(N/32) = 1024
  const int it   = blk & 31;
  const int bh   = blk >> 5;
  const int head = bh & (H_ - 1);
  const int b    = bh >> 2;
  const int i0   = it * 32;

  const float* dptr = dstv + (size_t)bh * N_;
  float lm = -1e30f;
#if GAT_ASYNC_LDS
  {
    // builtin signature: param0 = AS1 int* (global src), param1 = AS3 int* (LDS dst)
    typedef __attribute__((address_space(1))) int* as1i;
    typedef __attribute__((address_space(3))) int* as3i;
    for (int n = tid; n < N_; n += 256)
      __builtin_amdgcn_global_load_async_to_lds_b32(
          (as1i)(dptr + n), (as3i)(&ldsDst[n]), 0, 0);
#  if __has_builtin(__builtin_amdgcn_s_wait_asynccnt)
    __builtin_amdgcn_s_wait_asynccnt(0);
#  else
    asm volatile("s_wait_asynccnt 0" ::: "memory");
#  endif
    __syncthreads();
    for (int n = tid; n < N_; n += 256) lm = fmaxf(lm, ldsDst[n]);
  }
#else
  for (int n = tid; n < N_; n += 256) {
    float v = dptr[n]; ldsDst[n] = v; lm = fmaxf(lm, v);
  }
#endif
  red[tid] = lm;
  __syncthreads();
  for (int s = 128; s > 0; s >>= 1) {
    if (tid < s) red[tid] = fmaxf(red[tid], red[tid + s]);
    __syncthreads();
  }
  const float mdst = red[0];

  const int r  = i0 + isub * 16 + (lane & 15);
  const float si = srcv[(size_t)bh * N_ + r];
  float mi = si + mdst;  mi = mi > 0.f ? mi : 0.2f * mi;   // exact row max

  const int halfK = (lane >> 4) * 8;
  const int kl    = (lane >> 4) * 16;
  const bf16_t* hb_base = hbf + (size_t)bh * U_ * N_;

  // loop-carried h-tile pointers: advance by 32 elems (64B) per j-step instead
  // of recomputing full 64-bit addresses every iteration
  const bf16_t* hp[4];
#pragma unroll
  for (int ut = 0; ut < 4; ++ut)
    hp[ut] = hb_base + (size_t)(ug * 64 + ut * 16 + (lane & 15)) * N_ + kl;

  // per-lane base into the staged dst logits; unrolled e-offsets become
  // immediate ds_load offsets
  const float* dj = ldsDst + kl;

  f32x8 o[4] = {};
  float z = 0.f;

  for (int j0 = 0; j0 < N_; j0 += 32) {
    // issue h-tile loads first so they hide under the exp() work below
    bf16x16 hbm[4];
#pragma unroll
    for (int ut = 0; ut < 4; ++ut) {
      hbm[ut] = *(const bf16x16*)hp[ut];
      hp[ut] += 32;
    }

    // P tile directly in A-operand layout: elem e -> j = j0 + kl + e
    bf16x16 pa;
#pragma unroll
    for (int e = 0; e < 16; ++e) {
      float sc = si + dj[e];
      sc = sc > 0.f ? sc : 0.2f * sc;                      // leaky_relu(0.2)
      const float p = __expf(sc - mi);
      z += p;
      pa[e] = (bf16_t)p;
    }
    dj += 32;

#pragma unroll
    for (int ut = 0; ut < 4; ++ut)
      o[ut] = __builtin_amdgcn_wmma_f32_16x16x32_bf16(
          false, pa, false, hbm[ut], (short)0, o[ut], false, false);
  }

  // combine the two lane-halves' j-coverage for each row, share via LDS
  z += __shfl_xor(z, 16, 32);
  if (lane < 16 && ug == 0) ldsZ[isub * 16 + lane] = z;
  __syncthreads();

  float iz[8];
#pragma unroll
  for (int v = 0; v < 8; ++v) iz[v] = 1.0f / ldsZ[isub * 16 + halfK + v];

  // out[b, n, head*U + u]
  const int nb = i0 + isub * 16 + halfK;
#pragma unroll
  for (int ut = 0; ut < 4; ++ut) {
    const int u = ug * 64 + ut * 16 + (lane & 15);
    float* op = out + (((size_t)b * N_ + nb) * H_ + head) * U_ + u;
#pragma unroll
    for (int v = 0; v < 8; ++v)
      op[(size_t)v * H_ * U_] = o[ut][v] * iz[v];
  }
}

// ---------------------------------------------------------------------------
extern "C" void kernel_launch(void* const* d_in, const int* in_sizes, int n_in,
                              void* d_out, int out_size, void* d_ws, size_t ws_size,
                              hipStream_t stream) {
  (void)in_sizes; (void)n_in; (void)out_size; (void)ws_size;
  const float* x     = (const float*)d_in[0];   // [B,N,F]
  const float* W     = (const float*)d_in[1];   // [H,F,U]
  const float* a_src = (const float*)d_in[2];   // [H,U,1]
  const float* a_dst = (const float*)d_in[3];   // [H,U,1]
  float* out = (float*)d_out;                   // [B,N,H*U]

  // workspace: h bf16 [B*H][U][N] (16.8 MB) + src/dst logits (2*128 KB)
  bf16_t* hbf = (bf16_t*)d_ws;
  const size_t hElems = (size_t)B_ * H_ * U_ * N_;
  float* srcv = (float*)((char*)d_ws + hElems * sizeof(bf16_t));
  float* dstv = srcv + (size_t)B_ * H_ * N_;

  gat_proj<<<B_ * H_ * (N_ / 64), 256, 0, stream>>>(x, W, a_src, a_dst, hbf, srcv, dstv);
  gat_attn<<<B_ * H_ * (N_ / 32), 256, 0, stream>>>(hbf, srcv, dstv, out);
}
